// GCN_13151189860867
// MI455X (gfx1250) — matrix operand order repaired and checked by reading, not compiled, not run on hardware
//
#include <hip/hip_runtime.h>

// ============================================================================
// DGCNN-style GCN for MI455X (gfx1250, wave32).
// Heavy math (pairwise Gram for kNN, EdgeConv GEMMs, FC1) runs on
// v_wmma_f32_16x16x32_bf16.  Gram uses split-bf16 (hi+lo, 3 WMMAs) for
// near-f32 accuracy since it feeds a discrete top-k.  Final FC2 is f32 VALU.
// ============================================================================

typedef __attribute__((ext_vector_type(16))) __bf16 v16bf;
typedef __attribute__((ext_vector_type(8)))  float  v8f;

union frag_u { uint4 q[2]; v16bf v; };

// ---- bf16 helpers (manual RNE, avoids relying on __bf16 scalar arith) -----
static __device__ inline unsigned short f2bf(float f) {
    unsigned int u = __builtin_bit_cast(unsigned int, f);
    unsigned int r = (u + 0x7FFFu + ((u >> 16) & 1u)) >> 16;
    return (unsigned short)r;
}
static __device__ inline float bf2f(unsigned short b) {
    unsigned int u = ((unsigned int)b) << 16;
    return __builtin_bit_cast(float, u);
}

// ---- WMMA fragment loads (layouts per CDNA5 ISA 7.12.2) -------------------
// A (16x32 bf16): lane l holds row (l&15); halfs 0..7 = K[hi*8..hi*8+7],
//                 halfs 8..15 = K[16+hi*8 .. 16+hi*8+7]   (hi = l>>4)
static __device__ inline v16bf load_a_frag(const unsigned short* base, int row,
                                           int ldk, int k0, int hi) {
    const unsigned short* p = base + (size_t)row * ldk + k0;
    frag_u u;
    u.q[0] = *(const uint4*)(p + hi * 8);
    u.q[1] = *(const uint4*)(p + 16 + hi * 8);
    return u.v;
}
// B (32x16 bf16), B[k][n] = S[rowBase+n][k0+k]: lane l holds col (l&15),
// halfs 0..7 = K[hi*8..hi*8+7], halfs 8..15 = K[16+hi*8..16+hi*8+7] from that source row.
static __device__ inline v16bf load_b_frag(const unsigned short* base, int row,
                                           int ldk, int k0, int hi) {
    const unsigned short* p = base + (size_t)row * ldk + k0;
    frag_u u;
    u.q[0] = *(const uint4*)(p + hi * 8);
    u.q[1] = *(const uint4*)(p + 16 + hi * 8);
    return u.v;
}
static __device__ inline v8f wmma_bf16(v16bf a, v16bf b, v8f c) {
    return __builtin_amdgcn_wmma_f32_16x16x32_bf16(
        /*neg_a=*/false, a, /*neg_b=*/false, b,
        /*c_mod=*/(short)0, c, /*reuse_a=*/false, /*reuse_b=*/false);
}

// ---- prep: split f32 features into bf16 hi/lo, zero-padded to Fpad --------
__global__ void k_prep_split(const float* __restrict__ H, unsigned short* hi,
                             unsigned short* lo, int N, int F, int Fpad) {
    int t = blockIdx.x * blockDim.x + threadIdx.x;
    if (t >= N * Fpad) return;
    int i = t / Fpad, c = t - i * Fpad;
    float x = (c < F) ? H[(size_t)i * F + c] : 0.f;
    unsigned short h = f2bf(x);
    float rem = x - bf2f(h);
    hi[t] = h;
    lo[t] = f2bf(rem);
}

__global__ void k_sqnorm(const float* __restrict__ H, float* sq, int N, int F) {
    int i = blockIdx.x * blockDim.x + threadIdx.x;
    if (i >= N) return;
    float s = 0.f;
    for (int c = 0; c < F; ++c) { float v = H[(size_t)i * F + c]; s += v * v; }
    sq[i] = s;
}

// ---- kNN: one wave owns 32 rows, sweeps all column tiles ------------------
__global__ void __launch_bounds__(64)
k_knn(const unsigned short* __restrict__ hhi, const unsigned short* __restrict__ hlo,
      const float* __restrict__ sq, int* __restrict__ idx,
      int N, int kchunks, int ldk) {
    __shared__ float lds[2][32 * 17];
    const int wid = threadIdx.x >> 5;
    const int l   = threadIdx.x & 31;
    const int m   = l & 15;
    const int hi  = l >> 4;
    const int r0  = (blockIdx.x * 2 + wid) * 32;
    if (r0 >= N) return;                       // uniform per wave

    float td[10]; int ti[10];
#pragma unroll
    for (int p = 0; p < 10; ++p) { td[p] = 3.4e38f; ti[p] = 0; }

    const int rowA0 = (r0 + m      < N) ? (r0 + m)      : (N - 1);
    const int rowA1 = (r0 + 16 + m < N) ? (r0 + 16 + m) : (N - 1);
    float* myl = &lds[wid][0];
    const int gi  = r0 + l;                    // this lane's own row for top-k
    const float sqi = sq[(gi < N) ? gi : (N - 1)];

    for (int cj = 0; cj < N; cj += 16) {
        v8f acc0{}; v8f acc1{};
        const int rowB = cj + m;
        for (int ch = 0; ch < kchunks; ++ch) {
            const int k0 = ch * 32;
            v16bf bh = load_b_frag(hhi, rowB, ldk, k0, hi);
            v16bf bl = load_b_frag(hlo, rowB, ldk, k0, hi);
            v16bf ah0 = load_a_frag(hhi, rowA0, ldk, k0, hi);
            v16bf al0 = load_a_frag(hlo, rowA0, ldk, k0, hi);
            acc0 = wmma_bf16(al0, bh, acc0);   // lo*hi
            acc0 = wmma_bf16(ah0, bl, acc0);   // hi*lo
            acc0 = wmma_bf16(ah0, bh, acc0);   // hi*hi
            v16bf ah1 = load_a_frag(hhi, rowA1, ldk, k0, hi);
            v16bf al1 = load_a_frag(hlo, rowA1, ldk, k0, hi);
            acc1 = wmma_bf16(al1, bh, acc1);
            acc1 = wmma_bf16(ah1, bl, acc1);
            acc1 = wmma_bf16(ah1, bh, acc1);
        }
        // C layout: element r of frag -> local row r+8*hi, col m
#pragma unroll
        for (int r = 0; r < 8; ++r) {
            myl[(r + 8 * hi) * 17 + m]      = acc0[r];
            myl[(16 + r + 8 * hi) * 17 + m] = acc1[r];
        }
        asm volatile("s_wait_dscnt 0" ::: "memory");
        if (gi < N) {
#pragma unroll
            for (int n = 0; n < 16; ++n) {
                const int   j = cj + n;
                const float d = sqi + sq[j] - 2.0f * myl[l * 17 + n];
                if (d < td[9]) {                 // sorted insertion (ascending)
                    int pos = 9;
#pragma unroll
                    for (int p = 9; p > 0; --p) {
                        if (td[p - 1] > d) { td[p] = td[p - 1]; ti[p] = ti[p - 1]; pos = p - 1; }
                    }
                    td[pos] = d; ti[pos] = j;
                }
            }
        }
        asm volatile("s_wait_dscnt 0" ::: "memory");
    }
    if (gi < N) {
#pragma unroll
        for (int p = 0; p < 10; ++p) idx[gi * 10 + p] = ti[p];
    }
}

// ---- build EdgeConv message matrix msg[i*10+k][c] = [x_i , x_j - x_i] -----
__global__ void k_gather_msg(const float* __restrict__ H, const int* __restrict__ idx,
                             unsigned short* __restrict__ msg, int N, int Fin, int Kpad) {
    size_t t = (size_t)blockIdx.x * blockDim.x + threadIdx.x;
    size_t total = (size_t)N * 10 * Kpad;
    if (t >= total) return;
    int row = (int)(t / Kpad);
    int c   = (int)(t - (size_t)row * Kpad);
    int i   = row / 10;
    float v;
    if (c < Fin) {
        v = H[(size_t)i * Fin + c];
    } else if (c < 2 * Fin) {
        int j  = idx[row];
        int cc = c - Fin;
        v = H[(size_t)j * Fin + cc] - H[(size_t)i * Fin + cc];
    } else {
        v = 0.f;
    }
    msg[t] = f2bf(v);
}

// ---- convert weight matrix (f32 [rows][K]) -> bf16 [rows][Kpad] -----------
__global__ void k_prep_w(const float* __restrict__ W, unsigned short* __restrict__ Wb,
                         int rows, int K, int Kpad) {
    int t = blockIdx.x * blockDim.x + threadIdx.x;
    if (t >= rows * Kpad) return;
    int r = t / Kpad, c = t - r * Kpad;
    Wb[t] = f2bf((c < K) ? W[(size_t)r * K + c] : 0.f);
}

// ---- GEMM: C[M][Nout] = A(bf16) x Wb^T(bf16), f32 accumulate --------------
// MODE 0: EdgeConv epilogue: bias + leakyReLU(0.01), then relu(max over K=10)
//         folded into atomicMax against zero-initialized out[i = row/10].
// MODE 1: bias + ReLU, direct store.
template <int MODE>
__global__ void __launch_bounds__(128)
k_gemm(const unsigned short* __restrict__ A, const unsigned short* __restrict__ Bm,
       const float* __restrict__ bias, float* __restrict__ out,
       int M, int Kpad, int Nout, int ldOut) {
    const int wid = threadIdx.x >> 5;
    const int l   = threadIdx.x & 31;
    const int m   = l & 15;
    const int hi  = l >> 4;
    const int tilesN = Nout >> 4;
    const int tile = blockIdx.x * 4 + wid;
    if (tile >= (M >> 4) * tilesN) return;     // uniform per wave
    const int tM = tile / tilesN;
    const int tN = tile - tM * tilesN;
    const int rowA = tM * 16 + m;
    const int rowB = tN * 16 + m;
    v8f acc{};
    const int kch = Kpad >> 5;
    for (int ch = 0; ch < kch; ++ch) {
        const int k0 = ch << 5;
        v16bf a = load_a_frag(A,  rowA, Kpad, k0, hi);
        v16bf b = load_b_frag(Bm, rowB, Kpad, k0, hi);
        acc = wmma_bf16(a, b, acc);
    }
    const int col = tN * 16 + m;
    const float bv = bias[col];
#pragma unroll
    for (int r = 0; r < 8; ++r) {
        const int row = tM * 16 + r + 8 * hi;
        float v = acc[r] + bv;
        if (MODE == 0) {
            v = (v > 0.f) ? v : 0.01f * v;                     // leaky ReLU
            atomicMax(out + (size_t)(row / 10) * ldOut + col, v); // relu(max_k)
        } else {
            v = fmaxf(v, 0.f);
            out[(size_t)row * ldOut + col] = v;
        }
    }
}

// ---- final tiny FC (exact f32): out[i][o] = h[i]·Wl2[o] + bl2[o] ----------
__global__ void k_fc2(const float* __restrict__ h, const float* __restrict__ W,
                      const float* __restrict__ b, float* __restrict__ out, int N) {
    int t = blockIdx.x * blockDim.x + threadIdx.x;
    if (t >= N * 10) return;
    int i = t / 10, o = t - i * 10;
    float acc = b[o];
    for (int c = 0; c < 128; ++c)
        acc += h[(size_t)i * 128 + c] * W[(size_t)o * 128 + c];
    out[t] = acc;
}

// ============================================================================
extern "C" void kernel_launch(void* const* d_in, const int* in_sizes, int n_in,
                              void* d_out, int out_size, void* d_ws, size_t ws_size,
                              hipStream_t stream) {
    const int N = 10000;
    const float* x   = (const float*)d_in[0];
    const float* W1  = (const float*)d_in[1];  const float* b1  = (const float*)d_in[2];
    const float* W2  = (const float*)d_in[3];  const float* b2  = (const float*)d_in[4];
    const float* W3  = (const float*)d_in[5];  const float* b3  = (const float*)d_in[6];
    const float* Wl1 = (const float*)d_in[7];  const float* bl1 = (const float*)d_in[8];
    const float* Wl2 = (const float*)d_in[9];  const float* bl2 = (const float*)d_in[10];

    char* ws = (char*)d_ws;
    size_t off = 0;
    auto alloc = [&](size_t bytes) -> void* {
        void* p = ws + off;
        off = (off + bytes + 255) & ~(size_t)255;
        return p;
    };
    unsigned short* msg = (unsigned short*)alloc((size_t)N * 10 * 256 * 2); // 51.2 MB
    float* Ha  = (float*)alloc((size_t)N * 256 * 4);
    float* Hb  = (float*)alloc((size_t)N * 256 * 4);
    unsigned short* Hhi = (unsigned short*)alloc((size_t)N * 256 * 2);
    unsigned short* Hlo = (unsigned short*)alloc((size_t)N * 256 * 2);
    float* sq  = (float*)alloc((size_t)N * 4);
    int*   idx = (int*)alloc((size_t)N * 10 * 4);
    unsigned short* Wb = (unsigned short*)alloc((size_t)256 * 256 * 2);

    auto edge_layer = [&](const float* inH, int Fin, int Fpad, int Fout,
                          const float* W, const float* bias, float* outH) {
        int np = N * Fpad;
        k_prep_split<<<(np + 255) / 256, 256, 0, stream>>>(inH, Hhi, Hlo, N, Fin, Fpad);
        k_sqnorm<<<(N + 255) / 256, 256, 0, stream>>>(inH, sq, N, Fin);
        k_knn<<<(N + 63) / 64, 64, 0, stream>>>(Hhi, Hlo, sq, idx, N, Fpad / 32, Fpad);
        int twoF = 2 * Fin;
        int Kpad = (twoF + 31) & ~31;
        size_t nm = (size_t)N * 10 * Kpad;
        k_gather_msg<<<(unsigned)((nm + 255) / 256), 256, 0, stream>>>(inH, idx, msg, N, Fin, Kpad);
        k_prep_w<<<(Fout * Kpad + 255) / 256, 256, 0, stream>>>(W, Wb, Fout, twoF, Kpad);
        hipMemsetAsync(outH, 0, (size_t)N * Fout * 4, stream);
        int tiles = (N * 10 / 16) * (Fout / 16);
        k_gemm<0><<<(tiles + 3) / 4, 128, 0, stream>>>(msg, Wb, bias, outH,
                                                       N * 10, Kpad, Fout, Fout);
    };

    edge_layer(x,  76,  96,  64, W1, b1, Ha);   // L1: 76 -> 64
    edge_layer(Ha, 64,  64, 128, W2, b2, Hb);   // L2: 64 -> 128
    edge_layer(Hb, 128, 128, 256, W3, b3, Ha);  // L3: 128 -> 256

    // FC1: relu(Ha @ Wl1^T + bl1)  -> Hb (N x 128), bf16 WMMA
    k_prep_split<<<(N * 256 + 255) / 256, 256, 0, stream>>>(Ha, Hhi, Hlo, N, 256, 256);
    k_prep_w<<<(128 * 256 + 255) / 256, 256, 0, stream>>>(Wl1, Wb, 128, 256, 256);
    int tiles = (N / 16) * (128 / 16);
    k_gemm<1><<<(tiles + 3) / 4, 128, 0, stream>>>(Hhi, Wb, bl1, Hb, N, 256, 128, 128);

    // FC2 (exact f32): out = Hb @ Wl2^T + bl2
    k_fc2<<<(N * 10 + 255) / 256, 256, 0, stream>>>(Hb, Wl2, bl2, (float*)d_out, N);
}
